// Build_multimodal_fuse_head_54769422959015
// MI455X (gfx1250) — matrix operand order repaired and checked by reading, not compiled, not run on hardware
//
#include <hip/hip_runtime.h>
#include <hip/hip_bf16.h>

// ---------------------------------------------------------------------------
// MI455X (gfx1250) multimodal fuse head.
// v_wmma_f32_16x16x32_f16 for all K>=100 GEMM-like ops. Weights pre-packed to
// zero-padded f16 tiles in workspace -> LDS staging is pure b128 copies.
// One wave owns all M-tiles (B-fragment reuse); A-fragments preloaded in a
// batch (single dscnt wait per K-step); activation loads are branchless
// (clamped address + cndmask zero).
//
// Input flattening assumption (jax tree_flatten, sorted dict keys):
//   [0..22]  amm: cpb_b1,cpb_w1,cpb_w2,kpb,kpw,kw,logit_scale,n_b,n_g,p1b,
//            p1w,p2b,p2w,qpb,qpw,qw,sc_b,sc_g,scb,scw,vpb,vpw,vw
//   [23..58] mse[0..2]: b3,b4,bn1_b,bn1_s,bn3_b,bn3_s,se1,se2,w1,w2,w3,w4
//   [59..61] pos[0..2]; [62..64] sm_b,sm_s,sm_w; [65..67] x0,x1,x2
// Output: out (4,96,96,96) then de (4,36,384,384), fp32.
// ---------------------------------------------------------------------------

typedef __attribute__((ext_vector_type(16))) _Float16 v16h;
typedef __attribute__((ext_vector_type(8)))  _Float16 v8h;
typedef __attribute__((ext_vector_type(8)))  float    v8f;

#define BB   4
#define HH   384
#define WWp  384
#define HWI  (HH * WWp)      // 147456
#define C108 108
#define HP   96
#define HWP  (HP * HP)       // 9216
#define QS   48
#define QD   46
#define QL   (QD * QD)       // 2116
#define QHALF (QL / 2)       // 1058
#define VHALF (HWP / 2)      // 4608
#define EM   96

__device__ __forceinline__ v8f wmma32(v16h a, v16h b, v8f c) {
  return __builtin_amdgcn_wmma_f32_16x16x32_f16(false, a, false, b,
                                                (short)0, c, false, false);
}
__device__ __forceinline__ v16h cat8(v8h lo, v8h hi) {
  return __builtin_shufflevector(lo, hi, 0,1,2,3,4,5,6,7,8,9,10,11,12,13,14,15);
}
// A-fragment from an LDS row: elems 0..15 -> K=kbase+16*half+e.
__device__ __forceinline__ v16h a_frag_lds(const _Float16* row, int kbase, int half) {
  v8h lo = *(const v8h*)(row + kbase + 16 * half);
  v8h hi = *(const v8h*)(row + kbase + 16 * half + 8);
  return cat8(lo, hi);
}
__device__ __forceinline__ int clampi(int v, int lo, int hi) {
  return v < lo ? lo : (v > hi ? hi : v);
}

// ===================== weight pre-pack (fp32 -> padded f16) ================

// Flat (Cout,Cin) -> [MP][KP] f16, zero padded.
__global__ void k_pack_w(const float* __restrict__ W, _Float16* __restrict__ out,
                         int Cout, int Cin, int MP, int KP) {
  int i = blockIdx.x * blockDim.x + threadIdx.x;
  if (i >= MP * KP) return;
  int mm = i / KP, kk = i % KP;
  float v = W[(long)clampi(mm, 0, Cout - 1) * Cin + clampi(kk, 0, Cin - 1)];
  out[i] = (_Float16)((mm < Cout && kk < Cin) ? v : 0.f);
}

// Tapped (Cout,Cin,T) -> [T][MP][KP] f16, zero padded.
__global__ void k_pack_w_tap(const float* __restrict__ W, _Float16* __restrict__ out,
                             int Cout, int Cin, int T, int MP, int KP) {
  int i = blockIdx.x * blockDim.x + threadIdx.x;
  if (i >= T * MP * KP) return;
  int kk = i % KP; int mm = (i / KP) % MP; int tap = i / (KP * MP);
  float v = W[((long)clampi(mm, 0, Cout - 1) * Cin + clampi(kk, 0, Cin - 1)) * T + tap];
  out[i] = (_Float16)((mm < Cout && kk < Cin) ? v : 0.f);
}

// ========================= WMMA GEMM kernels ===============================

// 1x1 conv as GEMM. MODE 0:+scal 1:relu(+bias)+resid 2:+bias,GELU 3:+bias
template <int CIN, int COUT, int MODE, int HW>
__global__ void k_conv1x1_wmma(const float* __restrict__ in,
                               const _Float16* __restrict__ Wp,   // packed [MP][KP]
                               const float* __restrict__ bias,
                               const float* __restrict__ resid,
                               const float* __restrict__ scal,
                               float* __restrict__ out,
                               long in_bs, long out_bs, long res_bs, int ntiles) {
  constexpr int KSTEPS = (CIN + 31) / 32;
  constexpr int KP = KSTEPS * 32;
  constexpr int MT = (COUT + 15) / 16;
  constexpr int MP = MT * 16;
  __shared__ __attribute__((aligned(16))) _Float16 lw[MP * KP];
  {
    const v8h* s = (const v8h*)Wp;
    v8h* d = (v8h*)lw;
    for (int i = threadIdx.x; i < MP * KP / 8; i += blockDim.x) d[i] = s[i];
  }
  __syncthreads();

  int wave = threadIdx.x >> 5, lane = threadIdx.x & 31;
  int ntile = blockIdx.x * (blockDim.x >> 5) + wave;
  if (ntile >= ntiles) return;
  int half = (lane >> 4) & 1, l16 = lane & 15;
  long n = (long)ntile * 16 + l16;
  int b = (int)(n / HW), pix = (int)(n % HW);
  const float* inb = in + (long)b * in_bs + pix;

  v8f zero = {};
  v8f acc[MT];
#pragma unroll
  for (int mt = 0; mt < MT; ++mt) acc[mt] = zero;

#pragma unroll
  for (int ks = 0; ks < KSTEPS; ++ks) {
    const int kbase = ks * 32;
    float raw[16];
#pragma unroll
    for (int e = 0; e < 16; ++e) {
      int cin = kbase + e + 16 * half;
      int cc = (kbase + 31 < CIN) ? cin : clampi(cin, 0, CIN - 1);
      raw[e] = inb[(long)cc * HW];
    }
    v16h bf;
#pragma unroll
    for (int e = 0; e < 16; ++e) {
      int cin = kbase + e + 16 * half;
      bf[e] = (_Float16)((kbase + 31 < CIN) ? raw[e] : (cin < CIN ? raw[e] : 0.f));
    }
    v16h afs[MT];
#pragma unroll
    for (int mt = 0; mt < MT; ++mt)
      afs[mt] = a_frag_lds(&lw[(mt * 16 + l16) * KP], kbase, half);
#pragma unroll
    for (int mt = 0; mt < MT; ++mt)
      acc[mt] = wmma32(afs[mt], bf, acc[mt]);
  }
#pragma unroll
  for (int mt = 0; mt < MT; ++mt) {
#pragma unroll
    for (int r = 0; r < 8; ++r) {
      int cout = mt * 16 + r + 8 * half;
      if (cout < COUT) {
        float v = acc[mt][r];
        if (MODE == 0) v += scal[0];
        else if (MODE == 1)
          v = fmaxf(v + bias[cout], 0.f) + resid[(long)b * res_bs + (long)cout * HW + pix];
        else if (MODE == 2) { v += bias[cout]; v = 0.5f * v * (1.f + erff(v * 0.70710678118f)); }
        else v += bias[cout];
        out[(long)b * out_bs + (long)cout * HW + pix] = v;
      }
    }
  }
}

// de head: 3x3 conv 108->36 pad1 + BN + ReLU. Packed per-tap slices.
__global__ void k_conv3x3_fuse_wmma(const float* __restrict__ in,
                                    const _Float16* __restrict__ Wp,  // [9][48][128]
                                    const float* __restrict__ bns,
                                    const float* __restrict__ bnb,
                                    float* __restrict__ out) {
  __shared__ __attribute__((aligned(16))) _Float16 lw[48 * 128];
  int wave = threadIdx.x >> 5, lane = threadIdx.x & 31;
  int ntile = blockIdx.x * 4 + wave;          // exact grid, all waves active
  int half = (lane >> 4) & 1, l16 = lane & 15;
  long n = (long)ntile * 16 + l16;
  int b = (int)(n / HWI), pix = (int)(n % HWI);
  int y = pix / WWp, x = pix % WWp;
  const float* inb = in + (long)b * C108 * HWI;

  v8f zero = {};
  v8f acc[3] = {zero, zero, zero};
  for (int tap = 0; tap < 9; ++tap) {
    __syncthreads();
    {
      const v8h* s = (const v8h*)(Wp + (long)tap * 48 * 128);
      v8h* d = (v8h*)lw;
      for (int i = threadIdx.x; i < 48 * 128 / 8; i += 128) d[i] = s[i];
    }
    __syncthreads();
    int dy = tap / 3 - 1, dx = tap % 3 - 1;
    int yy = y + dy, xx = x + dx;
    bool valid = ((unsigned)yy < (unsigned)HH) && ((unsigned)xx < (unsigned)WWp);
    const float* pb = inb + (long)clampi(yy, 0, HH - 1) * WWp + clampi(xx, 0, WWp - 1);
    __builtin_prefetch(pb, 0, 1);
#pragma unroll
    for (int kc = 0; kc < 4; ++kc) {
      const int kbase = kc * 32;
      float raw[16];
#pragma unroll
      for (int e = 0; e < 16; ++e) {
        int cin = kbase + e + 16 * half;
        int cc = (kbase + 31 < C108) ? cin : clampi(cin, 0, C108 - 1);
        raw[e] = pb[(long)cc * HWI];
      }
      v16h bf;
#pragma unroll
      for (int e = 0; e < 16; ++e) {
        int cin = kbase + e + 16 * half;
        bool ok = valid && ((kbase + 31 < C108) || cin < C108);
        bf[e] = (_Float16)(ok ? raw[e] : 0.f);
      }
      v16h afs[3];
#pragma unroll
      for (int mt = 0; mt < 3; ++mt)
        afs[mt] = a_frag_lds(&lw[(mt * 16 + l16) * 128], kbase, half);
#pragma unroll
      for (int mt = 0; mt < 3; ++mt)
        acc[mt] = wmma32(afs[mt], bf, acc[mt]);
    }
  }
#pragma unroll
  for (int mt = 0; mt < 3; ++mt) {
#pragma unroll
    for (int r = 0; r < 8; ++r) {
      int cout = mt * 16 + r + 8 * half;
      if (cout < 36)
        out[((long)b * 36 + cout) * HWI + pix] =
            fmaxf(acc[mt][r] * bns[cout] + bnb[cout], 0.f);
    }
  }
}

// shortcut conv 4x4 s4 108->96 + bias. Packed per-tap slices.
__global__ void k_conv4x4s4_wmma(const float* __restrict__ in,
                                 const _Float16* __restrict__ Wp,  // [16][96][128]
                                 const float* __restrict__ bias,
                                 float* __restrict__ out) {
  __shared__ __attribute__((aligned(16))) _Float16 lw[96 * 128];
  int wave = threadIdx.x >> 5, lane = threadIdx.x & 31;
  int ntile = blockIdx.x * 4 + wave;
  int half = (lane >> 4) & 1, l16 = lane & 15;
  long n = (long)ntile * 16 + l16;
  int b = (int)(n / HWP), pp = (int)(n % HWP);
  int py = pp / HP, px = pp % HP;
  const float* inb = in + (long)b * C108 * HWI;

  v8f zero = {};
  v8f acc[6] = {zero, zero, zero, zero, zero, zero};
  for (int tap = 0; tap < 16; ++tap) {
    __syncthreads();
    {
      const v8h* s = (const v8h*)(Wp + (long)tap * 96 * 128);
      v8h* d = (v8h*)lw;
      for (int i = threadIdx.x; i < 96 * 128 / 8; i += 128) d[i] = s[i];
    }
    __syncthreads();
    int dy = tap >> 2, dx = tap & 3;
    const float* pb = inb + (long)(py * 4 + dy) * WWp + (px * 4 + dx);
    __builtin_prefetch(pb, 0, 1);
#pragma unroll
    for (int kc = 0; kc < 4; ++kc) {
      const int kbase = kc * 32;
      float raw[16];
#pragma unroll
      for (int e = 0; e < 16; ++e) {
        int cin = kbase + e + 16 * half;
        int cc = (kbase + 31 < C108) ? cin : clampi(cin, 0, C108 - 1);
        raw[e] = pb[(long)cc * HWI];
      }
      v16h bf;
#pragma unroll
      for (int e = 0; e < 16; ++e) {
        int cin = kbase + e + 16 * half;
        bool ok = (kbase + 31 < C108) || cin < C108;
        bf[e] = (_Float16)(ok ? raw[e] : 0.f);
      }
      v16h afs[6];
#pragma unroll
      for (int mt = 0; mt < 6; ++mt)
        afs[mt] = a_frag_lds(&lw[(mt * 16 + l16) * 128], kbase, half);
#pragma unroll
      for (int mt = 0; mt < 6; ++mt)
        acc[mt] = wmma32(afs[mt], bf, acc[mt]);
    }
  }
#pragma unroll
  for (int mt = 0; mt < 6; ++mt) {
#pragma unroll
    for (int r = 0; r < 8; ++r) {
      int cout = mt * 16 + r + 8 * half;
      out[((long)b * EM + cout) * HWP + pp] = acc[mt][r] + bias[cout];
    }
  }
}

// sim = (qn.kn^T)*ls + bias per (b,h): M=N=108, K=1058; K-slices in LDS.
__global__ void k_gemm_sim_wmma(const float* __restrict__ q,
                                const float* __restrict__ k,
                                const float* __restrict__ biasTab,
                                const float* __restrict__ lsp,
                                float* __restrict__ sim) {
  __shared__ __attribute__((aligned(16))) _Float16 lq[128 * 32];
  __shared__ __attribute__((aligned(16))) _Float16 lk[128 * 32];
  int wave = threadIdx.x >> 5, lane = threadIdx.x & 31;
  int ntile = blockIdx.x * 4 + wave;          // 0..7 (7 valid)
  int bh = blockIdx.y, b = bh >> 1, h = bh & 1;
  int half = (lane >> 4) & 1, l16 = lane & 15;
  const float* qb = q + (long)b * C108 * QL + h * QHALF;
  const float* kb = k + (long)b * C108 * QL + h * QHALF;

  v8f zero = {};
  v8f acc[7];
#pragma unroll
  for (int mt = 0; mt < 7; ++mt) acc[mt] = zero;

  for (int ks = 0; ks < 34; ++ks) {
    int kbase = ks * 32;
    __syncthreads();
    for (int i = threadIdx.x; i < 128 * 32; i += 128) {
      int row = i >> 5, kk = kbase + (i & 31);
      long idx = (long)clampi(row, 0, C108 - 1) * QL + clampi(kk, 0, QHALF - 1);
      bool ok = (row < C108) && (kk < QHALF);
      float vq = qb[idx], vk = kb[idx];
      lq[i] = (_Float16)(ok ? vq : 0.f);
      lk[i] = (_Float16)(ok ? vk : 0.f);
    }
    __syncthreads();
    int col = (ntile < 7 ? ntile : 6) * 16 + l16;
    v8h blo = *(const v8h*)(&lk[col * 32 + 16 * half]);
    v8h bhi = *(const v8h*)(&lk[col * 32 + 16 * half + 8]);
    v16h bf = cat8(blo, bhi);
    v16h afs[7];
#pragma unroll
    for (int mt = 0; mt < 7; ++mt)
      afs[mt] = a_frag_lds(&lq[(mt * 16 + l16) * 32], 0, half);
#pragma unroll
    for (int mt = 0; mt < 7; ++mt)
      acc[mt] = wmma32(afs[mt], bf, acc[mt]);
  }
  if (ntile >= 7) return;
  float ls = expf(fminf(lsp[h], 4.60517019f));
#pragma unroll
  for (int mt = 0; mt < 7; ++mt) {
#pragma unroll
    for (int r = 0; r < 8; ++r) {
      int i = mt * 16 + r + 8 * half;
      int j = ntile * 16 + l16;
      if (i < C108 && j < C108)
        sim[(((long)b * 2 + h) * C108 + i) * C108 + j] =
            acc[mt][r] * ls + biasTab[((long)h * C108 + i) * C108 + j];
    }
  }
}

// out = sim' @ v per (b,h): M=108, K=108, N=4608; sim staged whole in LDS.
__global__ void k_gemm_av_wmma(const float* __restrict__ sim,
                               const float* __restrict__ v,
                               float* __restrict__ outp) {
  __shared__ __attribute__((aligned(16))) _Float16 ls[112 * 128];
  int bh = blockIdx.y, b = bh >> 1, h = bh & 1;
  const float* sb = sim + (long)bh * C108 * C108;
  for (int i = threadIdx.x; i < 112 * 128; i += 128) {
    int mm = i >> 7, kk = i & 127;
    float vv = sb[(long)clampi(mm, 0, C108 - 1) * C108 + clampi(kk, 0, C108 - 1)];
    ls[i] = (_Float16)((mm < C108 && kk < C108) ? vv : 0.f);
  }
  __syncthreads();

  int wave = threadIdx.x >> 5, lane = threadIdx.x & 31;
  int ntile = blockIdx.x * 4 + wave;          // 0..287 exact
  int half = (lane >> 4) & 1, l16 = lane & 15;
  const float* vb = v + (long)b * C108 * HWP + h * VHALF;
  int d = ntile * 16 + l16;

  v8f zero = {};
  v8f acc[7];
#pragma unroll
  for (int mt = 0; mt < 7; ++mt) acc[mt] = zero;

#pragma unroll
  for (int ks = 0; ks < 4; ++ks) {
    const int kbase = ks * 32;
    float raw[16];
#pragma unroll
    for (int e = 0; e < 16; ++e) {
      int kk = kbase + e + 16 * half;
      int kcl = (kbase + 31 < C108) ? kk : clampi(kk, 0, C108 - 1);
      raw[e] = vb[(long)kcl * HWP + d];
    }
    v16h bf;
#pragma unroll
    for (int e = 0; e < 16; ++e) {
      int kk = kbase + e + 16 * half;
      bool ok = (kbase + 31 < C108) || kk < C108;
      bf[e] = (_Float16)(ok ? raw[e] : 0.f);
    }
    v16h afs[7];
#pragma unroll
    for (int mt = 0; mt < 7; ++mt)
      afs[mt] = a_frag_lds(&ls[(mt * 16 + l16) * 128], kbase, half);
#pragma unroll
    for (int mt = 0; mt < 7; ++mt)
      acc[mt] = wmma32(afs[mt], bf, acc[mt]);
  }
#pragma unroll
  for (int mt = 0; mt < 7; ++mt) {
#pragma unroll
    for (int r = 0; r < 8; ++r) {
      int cout = mt * 16 + r + 8 * half;
      if (cout < C108)
        outp[((long)b * C108 + cout) * HWP + h * VHALF + d] = acc[mt][r];
    }
  }
}

// ========================= VALU / LDS kernels ==============================

__global__ void k_conv3x3_c3(const float* __restrict__ x, const float* __restrict__ w,
                             const float* __restrict__ s, const float* __restrict__ bsh,
                             float* __restrict__ y) {
  __shared__ float wl[36 * 27];
  __shared__ float sl[36], bl[36];
  for (int i = threadIdx.x; i < 36 * 27; i += blockDim.x) wl[i] = w[i];
  if (threadIdx.x < 36) { sl[threadIdx.x] = s[threadIdx.x]; bl[threadIdx.x] = bsh[threadIdx.x]; }
  __syncthreads();
  long gid = (long)blockIdx.x * blockDim.x + threadIdx.x;
  if (gid >= (long)BB * 36 * HWI) return;
  int pix = (int)(gid % HWI); long t2 = gid / HWI;
  int co = (int)(t2 % 36); int b = (int)(t2 / 36);
  int y0 = pix / WWp, x0 = pix % WWp;
  float acc = 0.f;
  for (int ci = 0; ci < 3; ++ci)
    for (int dy = -1; dy <= 1; ++dy)
      for (int dx = -1; dx <= 1; ++dx) {
        int yy = y0 + dy, xx = x0 + dx;
        if ((unsigned)yy < (unsigned)HH && (unsigned)xx < (unsigned)WWp)
          acc += wl[(co * 3 + ci) * 9 + (dy + 1) * 3 + (dx + 1)] *
                 x[((long)b * 3 + ci) * HWI + (long)yy * WWp + xx];
      }
  y[gid] = fmaxf(acc * sl[co] + bl[co], 0.f);
}

__global__ void k_dwgconv3x3(const float* __restrict__ x, const float* __restrict__ w,
                             const float* __restrict__ b3, const float* __restrict__ s,
                             const float* __restrict__ bsh, float* __restrict__ y) {
  __shared__ float wl[18 * 54];
  __shared__ float bb[18], sl[18], bl[18];
  for (int i = threadIdx.x; i < 18 * 54; i += blockDim.x) wl[i] = w[i];
  if (threadIdx.x < 18) {
    bb[threadIdx.x] = b3[threadIdx.x]; sl[threadIdx.x] = s[threadIdx.x];
    bl[threadIdx.x] = bsh[threadIdx.x];
  }
  __syncthreads();
  long gid = (long)blockIdx.x * blockDim.x + threadIdx.x;
  if (gid >= (long)BB * 18 * HWI) return;
  int pix = (int)(gid % HWI); long t2 = gid / HWI;
  int co = (int)(t2 % 18); int b = (int)(t2 / 18);
  int g = co / 6;
  int y0 = pix / WWp, x0 = pix % WWp;
  float acc = 0.f;
  for (int j = 0; j < 6; ++j)
    for (int dy = -1; dy <= 1; ++dy)
      for (int dx = -1; dx <= 1; ++dx) {
        int yy = y0 + dy, xx = x0 + dx;
        if ((unsigned)yy < (unsigned)HH && (unsigned)xx < (unsigned)WWp)
          acc += wl[(co * 6 + j) * 9 + (dy + 1) * 3 + (dx + 1)] *
                 x[((long)b * 18 + g * 6 + j) * HWI + (long)yy * WWp + xx];
      }
  y[gid] = (acc + bb[co]) * sl[co] + bl[co];
}

__global__ void k_chan_mean(const float* __restrict__ z, float* __restrict__ mean, int cb) {
  __shared__ float red[256];
  int c = blockIdx.x, b = blockIdx.y;
  const float* p = z + ((long)b * C108 + cb + c) * HWI;
  float s = 0.f;
  for (int i = threadIdx.x; i < HWI; i += 256) s += p[i];
  red[threadIdx.x] = s; __syncthreads();
  for (int st = 128; st > 0; st >>= 1) {
    if (threadIdx.x < st) red[threadIdx.x] += red[threadIdx.x + st];
    __syncthreads();
  }
  if (threadIdx.x == 0) mean[b * 36 + c] = red[0] / (float)HWI;
}

__global__ void k_se_coeff(const float* __restrict__ mean, const float* __restrict__ se1,
                           const float* __restrict__ se2, float* __restrict__ coef) {
  __shared__ float m[36], mid[2];
  int b = blockIdx.x, t = threadIdx.x;
  if (t < 36) m[t] = mean[b * 36 + t];
  __syncthreads();
  if (t < 2) {
    float a = 0.f;
    for (int c = 0; c < 36; ++c) a += se1[t * 36 + c] * m[c];
    mid[t] = fminf(fmaxf(a, 0.f), 6.f);
  }
  __syncthreads();
  if (t < 36) {
    float a = se2[t * 2 + 0] * mid[0] + se2[t * 2 + 1] * mid[1];
    coef[b * 36 + t] = 1.f / (1.f + expf(-a));
  }
}

__global__ void k_se_scale(float* __restrict__ fuse, const float* __restrict__ coef, int cb) {
  long gid = (long)blockIdx.x * blockDim.x + threadIdx.x;
  if (gid >= (long)BB * 36 * HWI) return;
  int pix = (int)(gid % HWI); long t2 = gid / HWI;
  int c = (int)(t2 % 36); int b = (int)(t2 / 36);
  fuse[((long)b * C108 + cb + c) * HWI + pix] *= coef[b * 36 + c];
}

__global__ void k_qkpool(const float* __restrict__ fuse, const float* __restrict__ w,
                         float* __restrict__ outp) {
  __shared__ float wr[36];
  __shared__ float red[64];
  int c = blockIdx.y, b = blockIdx.z;
  int sp = blockIdx.x;
  int ph = sp / QS, pw = sp % QS;
  int t = threadIdx.x;
  if (t < 36) wr[t] = w[c * 36 + t];
  __syncthreads();
  int wy = t >> 3, wx = t & 7;
  int yy = ph * 8 + wy, xx = pw * 8 + wx;
  int g = c / 36;
  const float* p = fuse + ((long)b * C108 + g * 36) * HWI + (long)yy * WWp + xx;
  float acc = 0.f;
  for (int j = 0; j < 36; ++j) acc += wr[j] * p[(long)j * HWI];
  red[t] = acc; __syncthreads();
  for (int st = 32; st > 0; st >>= 1) {
    if (t < st) red[t] = fmaxf(red[t], red[t + st]);
    __syncthreads();
  }
  if (t == 0) outp[((long)b * C108 + c) * (QS * QS) + sp] = red[0];
}

__global__ void k_dw3x3v(const float* __restrict__ inp, const float* __restrict__ w,
                         const float* __restrict__ bias, float* __restrict__ outp) {
  long gid = (long)blockIdx.x * blockDim.x + threadIdx.x;
  if (gid >= (long)BB * C108 * QL) return;
  int ox = (int)(gid % QD); long t2 = gid / QD;
  int oy = (int)(t2 % QD); t2 /= QD;
  int c = (int)(t2 % C108); int b = (int)(t2 / C108);
  const float* p = inp + ((long)b * C108 + c) * (QS * QS);
  float acc = bias[c];
  for (int dy = 0; dy < 3; ++dy)
    for (int dx = 0; dx < 3; ++dx)
      acc += w[c * 9 + dy * 3 + dx] * p[(oy + dy) * QS + (ox + dx)];
  outp[((long)b * C108 + c) * QL + oy * QD + ox] = acc;
}

__global__ void k_l2half(float* __restrict__ q) {
  __shared__ float red[128];
  int id = blockIdx.x;
  int h = id & 1; int c = (id >> 1) % C108; int b = id / (2 * C108);
  float* p = q + ((long)b * C108 + c) * QL + h * QHALF;
  float s = 0.f;
  for (int i = threadIdx.x; i < QHALF; i += 128) { float v = p[i]; s += v * v; }
  red[threadIdx.x] = s; __syncthreads();
  for (int st = 64; st > 0; st >>= 1) {
    if (threadIdx.x < st) red[threadIdx.x] += red[threadIdx.x + st];
    __syncthreads();
  }
  float inv = 1.f / fmaxf(sqrtf(red[0]), 1e-12f);
  for (int i = threadIdx.x; i < QHALF; i += 128) p[i] *= inv;
}

__global__ void k_vpatch(const float* __restrict__ fuse, const float* __restrict__ vw,
                         const float* __restrict__ vpw, const float* __restrict__ vpb,
                         float* __restrict__ outp) {
  __shared__ float wr[36];
  __shared__ float tp[16];
  int c = blockIdx.y, b = blockIdx.z;
  int t = threadIdx.x;
  if (t < 36) wr[t] = vw[c * 36 + t];
  if (t < 16) tp[t] = vpw[c * 16 + t];
  __syncthreads();
  int pp = blockIdx.x * 256 + t;
  int py = pp / HP, px = pp % HP;
  int g = c / 36;
  const float* base = fuse + ((long)b * C108 + g * 36) * HWI;
  float acc = vpb[c];
  for (int dy = 0; dy < 4; ++dy)
    for (int dx = 0; dx < 4; ++dx) {
      const float* p = base + (long)(py * 4 + dy) * WWp + (px * 4 + dx);
      float s = 0.f;
      for (int j = 0; j < 36; ++j) s += wr[j] * p[(long)j * HWI];
      acc += tp[dy * 4 + dx] * s;
    }
  outp[((long)b * C108 + c) * HWP + pp] = acc;
}

__global__ void k_cpb(const float* __restrict__ w1, const float* __restrict__ b1,
                      const float* __restrict__ w2, float* __restrict__ biasTab) {
  int gid = blockIdx.x * blockDim.x + threadIdx.x;
  if (gid >= 2 * C108 * C108) return;
  int j = gid % C108; int i = (gid / C108) % C108; int h = gid / (C108 * C108);
  float r = (float)(j - i) / 107.f * 8.f;
  float sgn = (r > 0.f) ? 1.f : ((r < 0.f) ? -1.f : 0.f);
  float val = sgn * log2f(fabsf(r) + 1.f) * (1.f / 3.f);
  float o = 0.f;
  for (int kk = 0; kk < 48; ++kk) {
    float hk = fmaxf(w1[kk] * val + b1[kk], 0.f);
    o += hk * w2[h * 48 + kk];
  }
  biasTab[gid] = 1.f / (1.f + expf(-o));
}

__global__ void k_softmax_double(float* __restrict__ sim) {
  __shared__ float red[128];
  int t = threadIdx.x;
  float* p = sim + (long)blockIdx.x * C108;
  float x = (t < C108) ? p[t] : -3.4e38f;
  red[t] = x; __syncthreads();
  for (int st = 64; st > 0; st >>= 1) { if (t < st) red[t] = fmaxf(red[t], red[t + st]); __syncthreads(); }
  float mx = red[0]; __syncthreads();
  float e1 = (t < C108) ? expf(x - mx) : 0.f;
  red[t] = e1; __syncthreads();
  for (int st = 64; st > 0; st >>= 1) { if (t < st) red[t] += red[t + st]; __syncthreads(); }
  float s1 = red[0]; __syncthreads();
  float y = (t < C108) ? (1.f - e1 / s1) : -3.4e38f;
  red[t] = y; __syncthreads();
  for (int st = 64; st > 0; st >>= 1) { if (t < st) red[t] = fmaxf(red[t], red[t + st]); __syncthreads(); }
  float mx2 = red[0]; __syncthreads();
  float e2 = (t < C108) ? expf(y - mx2) : 0.f;
  red[t] = e2; __syncthreads();
  for (int st = 64; st > 0; st >>= 1) { if (t < st) red[t] += red[t + st]; __syncthreads(); }
  float s2 = red[0];
  if (t < C108) p[t] = e2 / s2;
}

__global__ void k_ln_combine(const float* __restrict__ a, const float* __restrict__ bsrc,
                             const float* __restrict__ ng, const float* __restrict__ nb,
                             const float* __restrict__ sg, const float* __restrict__ sb,
                             float* __restrict__ outp) {
  int gid = blockIdx.x * blockDim.x + threadIdx.x;
  if (gid >= BB * HWP) return;
  int b = gid / HWP, pp = gid % HWP;
  const float* pa = a + (long)b * EM * HWP + pp;
  const float* pb = bsrc + (long)b * EM * HWP + pp;
  float s1 = 0.f, q1 = 0.f, s2 = 0.f, q2 = 0.f;
  for (int c = 0; c < EM; ++c) {
    float v = pa[(long)c * HWP]; s1 += v; q1 += v * v;
    float u = pb[(long)c * HWP]; s2 += u; q2 += u * u;
  }
  float m1 = s1 / EM, m2 = s2 / EM;
  float iv1 = rsqrtf(q1 / EM - m1 * m1 + 1e-5f);
  float iv2 = rsqrtf(q2 / EM - m2 * m2 + 1e-5f);
  for (int c = 0; c < EM; ++c) {
    float v1 = (pa[(long)c * HWP] - m1) * iv1 * ng[c] + nb[c];
    float v2 = (pb[(long)c * HWP] - m2) * iv2 * sg[c] + sb[c];
    outp[(long)b * EM * HWP + (long)c * HWP + pp] = v1 + v2;
  }
}

// ============================== launcher ===================================

extern "C" void kernel_launch(void* const* d_in, const int* in_sizes, int n_in,
                              void* d_out, int out_size, void* d_ws, size_t ws_size,
                              hipStream_t stream) {
  (void)in_sizes; (void)n_in; (void)out_size; (void)ws_size;
  const float* cpb_b1 = (const float*)d_in[0];
  const float* cpb_w1 = (const float*)d_in[1];
  const float* cpb_w2 = (const float*)d_in[2];
  const float* kpb = (const float*)d_in[3];
  const float* kpw = (const float*)d_in[4];
  const float* kw  = (const float*)d_in[5];
  const float* lsc = (const float*)d_in[6];
  const float* n_b = (const float*)d_in[7];
  const float* n_g = (const float*)d_in[8];
  const float* p1b = (const float*)d_in[9];
  const float* p1w = (const float*)d_in[10];
  const float* p2b = (const float*)d_in[11];
  const float* p2w = (const float*)d_in[12];
  const float* qpb = (const float*)d_in[13];
  const float* qpw = (const float*)d_in[14];
  const float* qw  = (const float*)d_in[15];
  const float* sc_b = (const float*)d_in[16];
  const float* sc_g = (const float*)d_in[17];
  const float* scb = (const float*)d_in[18];
  const float* scw = (const float*)d_in[19];
  const float* vpb = (const float*)d_in[20];
  const float* vpw = (const float*)d_in[21];
  const float* vw  = (const float*)d_in[22];
  const float* mse[3][12];
  for (int i = 0; i < 3; ++i)
    for (int j = 0; j < 12; ++j)
      mse[i][j] = (const float*)d_in[23 + i * 12 + j];
  const float* pos[3] = {(const float*)d_in[59], (const float*)d_in[60], (const float*)d_in[61]};
  const float* sm_b = (const float*)d_in[62];
  const float* sm_s = (const float*)d_in[63];
  const float* sm_w = (const float*)d_in[64];
  const float* xin[3] = {(const float*)d_in[65], (const float*)d_in[66], (const float*)d_in[67]};

  // ---- workspace: packed f16 weights first (16B aligned), then fp32 bufs ----
  _Float16* Hf = (_Float16*)d_ws;
  size_t hoff = 0;
  auto carveh = [&](size_t nel) { _Float16* p = Hf + hoff; hoff += nel; return p; };
  _Float16* pk_w2 = carveh(32 * 64);            // <36,18>
  _Float16* pk_w4 = carveh(48 * 32);            // <18,36>
  _Float16* pk_p1 = carveh(112 * 128);          // <108,108>
  _Float16* pk_p2 = carveh(96 * 128);           // <108,96>
  _Float16* pk_de = carveh((size_t)9 * 48 * 128);
  _Float16* pk_sc = carveh((size_t)16 * 96 * 128);
  hoff = (hoff + 7) & ~(size_t)7;

  float* Wf = (float*)(Hf + hoff);
  size_t off = 0;
  auto carve = [&](size_t nel) { float* p = Wf + off; off += nel; return p; };
  float* ws_fuse = carve((size_t)BB * C108 * HWI);
  float* ws_y1   = carve((size_t)BB * 36 * HWI);
  float* ws_t    = carve((size_t)BB * 18 * HWI);
  float* ws_u    = carve((size_t)BB * 18 * HWI);
  float* ws_qp   = carve((size_t)BB * C108 * QS * QS);
  float* ws_kp   = carve((size_t)BB * C108 * QS * QS);
  float* ws_q    = carve((size_t)BB * C108 * QL);
  float* ws_k    = carve((size_t)BB * C108 * QL);
  float* ws_v    = carve((size_t)BB * C108 * HWP);
  float* ws_sim  = carve((size_t)BB * 2 * C108 * C108);
  float* ws_bias = carve((size_t)2 * C108 * C108);
  float* ws_attn = carve((size_t)BB * C108 * HWP);
  float* ws_p1   = carve((size_t)BB * C108 * HWP);
  float* ws_p2   = carve((size_t)BB * EM * HWP);
  float* ws_scp  = carve((size_t)BB * EM * HWP);
  float* ws_sem  = carve((size_t)BB * 36);
  float* ws_sec  = carve((size_t)BB * 36);

  float* out_main = (float*)d_out;
  float* out_de   = out_main + (size_t)BB * EM * HWP;

  const int ntI = BB * HWI / 16;   // 36864
  const int ntP = BB * HWP / 16;   // 2304

  // pack the branch-invariant weights once
  k_pack_w<<<(112 * 128 + 255) / 256, 256, 0, stream>>>(p1w, pk_p1, 108, 108, 112, 128);
  k_pack_w<<<(96 * 128 + 255) / 256, 256, 0, stream>>>(p2w, pk_p2, 96, 108, 96, 128);
  k_pack_w_tap<<<(9 * 48 * 128 + 255) / 256, 256, 0, stream>>>(sm_w, pk_de, 36, 108, 9, 48, 128);
  k_pack_w_tap<<<(16 * 96 * 128 + 255) / 256, 256, 0, stream>>>(scw, pk_sc, 96, 108, 16, 96, 128);

  for (int br = 0; br < 3; ++br) {
    const float* b3  = mse[br][0];  const float* b4  = mse[br][1];
    const float* bn1b = mse[br][2]; const float* bn1s = mse[br][3];
    const float* bn3b = mse[br][4]; const float* bn3s = mse[br][5];
    const float* se1 = mse[br][6];  const float* se2 = mse[br][7];
    const float* w1  = mse[br][8];  const float* w2  = mse[br][9];
    const float* w3  = mse[br][10]; const float* w4  = mse[br][11];

    long tot1 = (long)BB * 36 * HWI;
    k_conv3x3_c3<<<(int)((tot1 + 255) / 256), 256, 0, stream>>>(xin[br], w1, bn1s, bn1b, ws_y1);

    k_pack_w<<<(32 * 64 + 255) / 256, 256, 0, stream>>>(w2, pk_w2, 18, 36, 32, 64);
    k_conv1x1_wmma<36, 18, 0, HWI><<<ntI / 4, 128, 0, stream>>>(
        ws_y1, pk_w2, nullptr, nullptr, pos[br], ws_t,
        (long)36 * HWI, (long)18 * HWI, 0, ntI);

    long tot3 = (long)BB * 18 * HWI;
    k_dwgconv3x3<<<(int)((tot3 + 255) / 256), 256, 0, stream>>>(ws_t, w3, b3, bn3s, bn3b, ws_u);

    k_pack_w<<<(48 * 32 + 255) / 256, 256, 0, stream>>>(w4, pk_w4, 36, 18, 48, 32);
    k_conv1x1_wmma<18, 36, 1, HWI><<<ntI / 4, 128, 0, stream>>>(
        ws_u, pk_w4, b4, ws_y1, nullptr, ws_fuse + (size_t)br * 36 * HWI,
        (long)18 * HWI, (long)C108 * HWI, (long)36 * HWI, ntI);

    dim3 gm(36, BB);
    k_chan_mean<<<gm, 256, 0, stream>>>(ws_fuse, ws_sem, 36 * br);
    k_se_coeff<<<BB, 64, 0, stream>>>(ws_sem, se1, se2, ws_sec);
    long tot5 = (long)BB * 36 * HWI;
    k_se_scale<<<(int)((tot5 + 255) / 256), 256, 0, stream>>>(ws_fuse, ws_sec, 36 * br);
  }

  k_conv3x3_fuse_wmma<<<ntI / 4, 128, 0, stream>>>(ws_fuse, pk_de, sm_s, sm_b, out_de);

  dim3 gq(QS * QS, C108, BB);
  k_qkpool<<<gq, 64, 0, stream>>>(ws_fuse, qw, ws_qp);
  k_qkpool<<<gq, 64, 0, stream>>>(ws_fuse, kw, ws_kp);

  long totd = (long)BB * C108 * QL;
  k_dw3x3v<<<(int)((totd + 255) / 256), 256, 0, stream>>>(ws_qp, qpw, qpb, ws_q);
  k_dw3x3v<<<(int)((totd + 255) / 256), 256, 0, stream>>>(ws_kp, kpw, kpb, ws_k);

  k_l2half<<<BB * C108 * 2, 128, 0, stream>>>(ws_q);
  k_l2half<<<BB * C108 * 2, 128, 0, stream>>>(ws_k);

  dim3 gv(HWP / 256, C108, BB);
  k_vpatch<<<gv, 256, 0, stream>>>(ws_fuse, vw, vpw, vpb, ws_v);

  k_cpb<<<(2 * C108 * C108 + 255) / 256, 256, 0, stream>>>(cpb_w1, cpb_b1, cpb_w2, ws_bias);

  dim3 gs(2, BB * 2);
  k_gemm_sim_wmma<<<gs, 128, 0, stream>>>(ws_q, ws_k, ws_bias, lsc, ws_sim);

  k_softmax_double<<<BB * 2 * C108, 128, 0, stream>>>(ws_sim);

  dim3 ga(72, BB * 2);
  k_gemm_av_wmma<<<ga, 128, 0, stream>>>(ws_sim, ws_v, ws_attn);

  k_conv1x1_wmma<108, 108, 2, HWP><<<ntP / 4, 128, 0, stream>>>(
      ws_attn, pk_p1, p1b, nullptr, nullptr, ws_p1,
      (long)108 * HWP, (long)108 * HWP, 0, ntP);
  k_conv1x1_wmma<108, 96, 3, HWP><<<ntP / 4, 128, 0, stream>>>(
      ws_p1, pk_p2, p2b, nullptr, nullptr, ws_p2,
      (long)108 * HWP, (long)96 * HWP, 0, ntP);

  k_conv4x4s4_wmma<<<ntP / 4, 128, 0, stream>>>(ws_fuse, pk_sc, scb, ws_scp);

  k_ln_combine<<<(BB * HWP + 255) / 256, 256, 0, stream>>>(ws_p2, ws_scp, n_g, n_b,
                                                           sc_g, sc_b, out_main);
}
